// PhiMoEAttention_4990751998307
// MI455X (gfx1250) — matrix operand order repaired and checked by reading, not compiled, not run on hardware
//
#include <hip/hip_runtime.h>
#include <hip/hip_bf16.h>

// Sizes fixed by the reference
#define SEQ   4096
#define HID   4096
#define NQH   32
#define NKVH  8
#define DH    128
#define QKVN  6144          // NQ*D + 2*NKV*D
#define QSC   0.08838834764831845f   // 1/sqrt(128)
#define ROPE_LN 0.14391156831212787f // ln(10000)/64

typedef __bf16 bf16_t;
typedef bf16_t v16bf __attribute__((ext_vector_type(16)));
typedef float  v8f   __attribute__((ext_vector_type(8)));

union BF16Frag { v16bf v; uint4 q[2]; };

// Load a WMMA fragment as two 16-byte chunks (same contiguous K order for both A and B:
// each lane reads one 32B run at K = g*16 .. g*16+15)
static __device__ __forceinline__ v16bf frag_ld2(const bf16_t* p0, const bf16_t* p1) {
  BF16Frag f;
  f.q[0] = *(const uint4*)p0;
  f.q[1] = *(const uint4*)p1;
  return f.v;
}

static __device__ __forceinline__ v8f wmma_bf16(v16bf a, v16bf b, v8f c) {
  return __builtin_amdgcn_wmma_f32_16x16x32_bf16(false, a, false, b, (short)0, c,
                                                 false, false);
}

// ---------------- conversion kernels ----------------
__global__ void cvt_bf16(const float* __restrict__ in, bf16_t* __restrict__ out, int n) {
  int i = (blockIdx.x * blockDim.x + threadIdx.x) * 4;
  if (i >= n) return;
  float4 v = *(const float4*)(in + i);
  out[i + 0] = (bf16_t)v.x;
  out[i + 1] = (bf16_t)v.y;
  out[i + 2] = (bf16_t)v.z;
  out[i + 3] = (bf16_t)v.w;
}

// out[c*R + r] = bf16(in[r*C + c])  (row-major RxC -> transposed bf16 NxK)
__global__ void cvt_transpose(const float* __restrict__ in, bf16_t* __restrict__ out,
                              int R, int C) {
  size_t i = (size_t)blockIdx.x * blockDim.x + threadIdx.x;
  size_t total = (size_t)R * C;
  if (i >= total) return;
  int r = (int)(i / C);
  int c = (int)(i % C);
  out[(size_t)c * R + r] = (bf16_t)in[i];
}

// ---------------- GEMM: C(MxN,f32) = A(MxK,bf16 row-major) * B^T (B stored NxK bf16)
// One wave computes a 32x64 output tile: 2 A-fragments x 4 B-fragments = 8 WMMA/k-step.
__global__ __launch_bounds__(128)
void gemm_bf16_nt(const bf16_t* __restrict__ A, const bf16_t* __restrict__ B,
                  float* __restrict__ C, int M, int N, int K) {
  const int lane = threadIdx.x & 31;
  const int wid  = threadIdx.x >> 5;
  const int wg   = blockIdx.x * 4 + wid;
  const int mtiles = M >> 5;                  // 32-row tiles
  const int m0 = (wg % mtiles) << 5;
  const int n0 = (wg / mtiles) << 6;          // 64-col strip
  if (n0 >= N) return;
  const int half = lane >> 4;                 // g (half-wave)
  const int ln   = lane & 15;

  const bf16_t* arow0 = A + (size_t)(m0 + ln) * K;
  const bf16_t* arow1 = A + (size_t)(m0 + 16 + ln) * K;
  const bf16_t* brow0 = B + (size_t)(n0 + 0 * 16 + ln) * K;
  const bf16_t* brow1 = B + (size_t)(n0 + 1 * 16 + ln) * K;
  const bf16_t* brow2 = B + (size_t)(n0 + 2 * 16 + ln) * K;
  const bf16_t* brow3 = B + (size_t)(n0 + 3 * 16 + ln) * K;

  v8f acc[2][4];
#pragma unroll
  for (int mi = 0; mi < 2; mi++)
#pragma unroll
    for (int nt = 0; nt < 4; nt++) acc[mi][nt] = {};

#pragma unroll 2
  for (int k0 = 0; k0 < K; k0 += 32) {
    // WGP-scope speculative prefetch ~256B ahead on each stream
    // (locality 3 -> near-cache; the bf16 tiles are L2-resident, we want L0/WGP$)
    __builtin_prefetch(arow0 + k0 + 128, 0, 3);
    __builtin_prefetch(arow1 + k0 + 128, 0, 3);
    __builtin_prefetch(brow0 + k0 + 128, 0, 3);
    __builtin_prefetch(brow1 + k0 + 128, 0, 3);
    __builtin_prefetch(brow2 + k0 + 128, 0, 3);
    __builtin_prefetch(brow3 + k0 + 128, 0, 3);

    v16bf af0 = frag_ld2(arow0 + k0 + half * 16, arow0 + k0 + half * 16 + 8);
    v16bf af1 = frag_ld2(arow1 + k0 + half * 16, arow1 + k0 + half * 16 + 8);
    const int bo = k0 + half * 16;
    v16bf b0 = frag_ld2(brow0 + bo, brow0 + bo + 8);
    v16bf b1 = frag_ld2(brow1 + bo, brow1 + bo + 8);
    v16bf b2 = frag_ld2(brow2 + bo, brow2 + bo + 8);
    v16bf b3 = frag_ld2(brow3 + bo, brow3 + bo + 8);
    acc[0][0] = wmma_bf16(af0, b0, acc[0][0]);
    acc[0][1] = wmma_bf16(af0, b1, acc[0][1]);
    acc[0][2] = wmma_bf16(af0, b2, acc[0][2]);
    acc[0][3] = wmma_bf16(af0, b3, acc[0][3]);
    acc[1][0] = wmma_bf16(af1, b0, acc[1][0]);
    acc[1][1] = wmma_bf16(af1, b1, acc[1][1]);
    acc[1][2] = wmma_bf16(af1, b2, acc[1][2]);
    acc[1][3] = wmma_bf16(af1, b3, acc[1][3]);
  }

  // C/D layout: vgpr r, lane -> (M = r + 8*half, N = ln)
#pragma unroll
  for (int mi = 0; mi < 2; mi++) {
#pragma unroll
    for (int nt = 0; nt < 4; nt++) {
#pragma unroll
      for (int r = 0; r < 8; r++) {
        C[(size_t)(m0 + mi * 16 + r + 8 * half) * N + n0 + nt * 16 + ln] =
            acc[mi][nt][r];
      }
    }
  }
}

// ---------------- RoPE + split/layout kernel ----------------
// qkv row: [ q(32*128) | k(8*128) | v(8*128) ]
// Qb: [s][32][128] bf16 (roped); Kb: [t][8][128] bf16 (roped); Vb: [g][d][t] bf16
__global__ void rope_split(const int* __restrict__ positions,
                           const float* __restrict__ qkv,
                           bf16_t* __restrict__ Qb, bf16_t* __restrict__ Kb,
                           bf16_t* __restrict__ Vb) {
  const int s = blockIdx.x;
  const float pos = (float)positions[s];
  const float* row = qkv + (size_t)s * QKVN;

  for (int idx = threadIdx.x; idx < 40 * 64; idx += blockDim.x) {
    int h = idx >> 6;
    int i = idx & 63;
    const float* hp = (h < NQH) ? (row + h * DH) : (row + NQH * DH + (h - NQH) * DH);
    float x1 = hp[i];
    float x2 = hp[i + 64];
    float inv = __expf(-ROPE_LN * (float)i);
    float ang = pos * inv;
    float sn, cs;
    __sincosf(ang, &sn, &cs);
    float o1 = x1 * cs - x2 * sn;
    float o2 = x2 * cs + x1 * sn;
    if (h < NQH) {
      bf16_t* q = Qb + ((size_t)s * NQH + h) * DH;
      q[i] = (bf16_t)o1;
      q[i + 64] = (bf16_t)o2;
    } else {
      bf16_t* k = Kb + ((size_t)s * NKVH + (h - NQH)) * DH;
      k[i] = (bf16_t)o1;
      k[i + 64] = (bf16_t)o2;
    }
  }
  for (int idx = threadIdx.x; idx < NKVH * DH; idx += blockDim.x) {
    int g = idx >> 7;
    int d = idx & 127;
    Vb[(size_t)g * DH * SEQ + (size_t)d * SEQ + s] =
        (bf16_t)row[NQH * DH + NKVH * DH + g * DH + d];
  }
}

// ---------------- flash attention: one wave per (head, 16-row q-tile) ----------------
__global__ __launch_bounds__(128)
void flash_attn(const bf16_t* __restrict__ Qb, const bf16_t* __restrict__ Kb,
                const bf16_t* __restrict__ Vb, bf16_t* __restrict__ Ab) {
  __shared__ __align__(16) bf16_t lds[4][16 * 32];
  const int lane = threadIdx.x & 31;
  const int wid  = threadIdx.x >> 5;
  const int h    = blockIdx.x;
  const int qt   = blockIdx.y * 4 + wid;
  const int s0   = qt << 4;
  const int gkv  = h >> 2;   // GRP = 4
  const int half = lane >> 4;
  const int ln   = lane & 15;
  bf16_t* pl = &lds[wid][0];

  // Q fragments for d0 = 0,32,64,96 (A-matrix, contiguous K = g*16 .. g*16+15)
  const bf16_t* qrow = Qb + ((size_t)(s0 + ln) * NQH + h) * DH;
  v16bf qf[4];
#pragma unroll
  for (int i = 0; i < 4; i++)
    qf[i] = frag_ld2(qrow + i * 32 + half * 16, qrow + i * 32 + half * 16 + 8);

  v8f acc[8];
#pragma unroll
  for (int n = 0; n < 8; n++) acc[n] = {};
  float m_i[8], l_i[8];
#pragma unroll
  for (int r = 0; r < 8; r++) { m_i[r] = -3.0e38f; l_i[r] = 0.0f; }

  const int nch = ((s0 + 15) >> 5) + 1;     // causal: keys t <= s0+15
  for (int c = 0; c < nch; c++) {
    const int tb = c << 5;
    // prefetch next chunk's K rows (near-cache) while computing this one
    if (c + 1 < nch) {
      __builtin_prefetch(
          Kb + ((size_t)(tb + 32 + ln) * NKVH + gkv) * DH + half * 16, 0, 3);
      __builtin_prefetch(
          Kb + ((size_t)(tb + 48 + ln) * NKVH + gkv) * DH + half * 16, 0, 3);
    }
    // --- scores: two 16x16 tiles over 32 keys, K-dim = 128 ---
    v8f sc[2];
#pragma unroll
    for (int sub = 0; sub < 2; sub++) {
      const bf16_t* kb =
          Kb + ((size_t)(tb + sub * 16 + ln) * NKVH + gkv) * DH + half * 16;
      v8f t = {};
#pragma unroll
      for (int i = 0; i < 4; i++) {
        v16bf kf = frag_ld2(kb + i * 32, kb + i * 32 + 8);
        t = wmma_bf16(qf[i], kf, t);
      }
      sc[sub] = t;
    }
    // --- mask + scale + online softmax ---
#pragma unroll
    for (int r = 0; r < 8; r++) {
      const int srow = s0 + r + 8 * half;
      float v0 = sc[0][r] * QSC;
      float v1 = sc[1][r] * QSC;
      if (tb + ln > srow)      v0 = -3.0e38f;
      if (tb + 16 + ln > srow) v1 = -3.0e38f;
      float mx = fmaxf(v0, v1);
#pragma unroll
      for (int off = 1; off < 16; off <<= 1) mx = fmaxf(mx, __shfl_xor(mx, off, 32));
      float mn = fmaxf(m_i[r], mx);
      float alpha = __expf(m_i[r] - mn);
      float p0 = __expf(v0 - mn);
      float p1 = __expf(v1 - mn);
      float rs = p0 + p1;
#pragma unroll
      for (int off = 1; off < 16; off <<= 1) rs += __shfl_xor(rs, off, 32);
      l_i[r] = l_i[r] * alpha + rs;
      m_i[r] = mn;
#pragma unroll
      for (int n = 0; n < 8; n++) acc[n][r] *= alpha;
      // C-fragment (M=r+8*half, N=ln) -> LDS tile [M][t_local]
      pl[(r + 8 * half) * 32 + ln]      = (bf16_t)p0;
      pl[(r + 8 * half) * 32 + 16 + ln] = (bf16_t)p1;
    }
    asm volatile("s_wait_dscnt 0" ::: "memory");
    // re-read as A-fragment (M = ln, K contiguous at g*16)
    v16bf pf = frag_ld2(pl + ln * 32 + half * 16, pl + ln * 32 + half * 16 + 8);
    // --- PV: 8 N-tiles across D=128, V stored [g][d][t] so K(=t) is contiguous ---
#pragma unroll
    for (int n = 0; n < 8; n++) {
      const bf16_t* vb =
          Vb + (size_t)gkv * DH * SEQ + (size_t)(n * 16 + ln) * SEQ + tb + half * 16;
      v16bf vf = frag_ld2(vb, vb + 8);
      acc[n] = wmma_bf16(pf, vf, acc[n]);
    }
  }
  // epilogue: normalize and store bf16 attn-out row-major [s][h*128+d]
#pragma unroll
  for (int n = 0; n < 8; n++) {
#pragma unroll
    for (int r = 0; r < 8; r++) {
      Ab[(size_t)(s0 + r + 8 * half) * (NQH * DH) + h * DH + n * 16 + ln] =
          (bf16_t)(acc[n][r] / l_i[r]);
    }
  }
}

extern "C" void kernel_launch(void* const* d_in, const int* in_sizes, int n_in,
                              void* d_out, int out_size, void* d_ws, size_t ws_size,
                              hipStream_t stream) {
  const int*   positions = (const int*)d_in[0];
  const float* x         = (const float*)d_in[1];
  const float* wqkv      = (const float*)d_in[2];
  const float* wo        = (const float*)d_in[3];
  float* out = (float*)d_out;

  // workspace layout (needs ~208 MiB); later buffers reuse dead earlier ones
  char* ws = (char*)d_ws;
  bf16_t* Xb    = (bf16_t*)(ws);                         // 32 MiB  (dead after gemm1)
  bf16_t* WqkvT = (bf16_t*)(ws + (size_t)(32u  << 20));  // 48 MiB  (dead after gemm1)
  bf16_t* WoT   = (bf16_t*)(ws + (size_t)(80u  << 20));  // 32 MiB
  float*  QKVf  = (float*) (ws + (size_t)(112u << 20));  // 96 MiB
  bf16_t* Qb    = (bf16_t*)(ws + (size_t)(32u  << 20));  // reuse WqkvT (32 MiB)
  bf16_t* Kb    = (bf16_t*)(ws + (size_t)(64u  << 20));  // reuse WqkvT (8 MiB)
  bf16_t* Vb    = (bf16_t*)(ws + (size_t)(72u  << 20));  // reuse WqkvT (8 MiB)
  bf16_t* Ab    = (bf16_t*)(ws);                         // reuse Xb (32 MiB)

  // 1) convert inputs to bf16 (weights transposed to NxK for B-fragment loads)
  {
    int n = SEQ * HID;
    cvt_bf16<<<n / (256 * 4), 256, 0, stream>>>(x, Xb, n);
  }
  {
    size_t total = (size_t)HID * QKVN;
    cvt_transpose<<<(unsigned)(total / 256), 256, 0, stream>>>(wqkv, WqkvT, HID, QKVN);
  }
  {
    size_t total = (size_t)HID * HID;
    cvt_transpose<<<(unsigned)(total / 256), 256, 0, stream>>>(wo, WoT, HID, HID);
  }

  // 2) QKV GEMM: (4096x4096) x (4096x6144) -> f32
  {
    int waves = (SEQ / 32) * (QKVN / 64);
    gemm_bf16_nt<<<waves / 4, 128, 0, stream>>>(Xb, WqkvT, QKVf, SEQ, QKVN, HID);
  }

  // 3) RoPE + split into Qb/Kb/Vb (V transposed [g][d][t])
  rope_split<<<SEQ, 256, 0, stream>>>(positions, QKVf, Qb, Kb, Vb);

  // 4) causal GQA flash attention -> Ab bf16
  {
    dim3 grid(NQH, (SEQ / 16) / 4);
    flash_attn<<<grid, 128, 0, stream>>>(Qb, Kb, Vb, Ab);
  }

  // 5) output projection: (4096x4096) x (4096x4096) -> d_out f32
  {
    int waves = (SEQ / 32) * (HID / 64);
    gemm_bf16_nt<<<waves / 4, 128, 0, stream>>>(Ab, WoT, out, SEQ, HID, HID);
  }
}
